// Attention_86165633892881
// MI455X (gfx1250) — matrix operand order repaired and checked by reading, not compiled, not run on hardware
//
#include <hip/hip_runtime.h>
#include <stdint.h>

// ---------------------------------------------------------------------------
// Cross-attention (B=2, N=M=2048, C=1024, H=16, hd=64) for gfx1250 (MI455X).
// All GEMM-shaped work goes through v_wmma_f32_16x16x32_bf16 (wave32 WMMA).
// Q/K tile staging uses GLOBAL_LOAD_ASYNC_TO_LDS (ASYNCcnt) when available.
// ---------------------------------------------------------------------------

typedef __attribute__((ext_vector_type(16))) __bf16 bf16x16;
typedef __attribute__((ext_vector_type(8)))  float  v8f;

#if defined(__gfx1250__) &&                                         \
    __has_builtin(__builtin_amdgcn_global_load_async_to_lds_b128) && \
    __has_builtin(__builtin_amdgcn_s_wait_asynccnt)
#define HAVE_ASYNC_LDS 1
#else
#define HAVE_ASYNC_LDS 0
#endif

#if HAVE_ASYNC_LDS
// Builtin signature (from hipcc diagnostics): parameters are pointers to a
// GCC-vector 'int __attribute__((vector_size(16)))' in AS1 (global) / AS3 (LDS).
typedef int v4i32_async __attribute__((vector_size(16)));
#define AS_GLOBAL_V4I(p) ((__attribute__((address_space(1))) v4i32_async*)(p))
#define AS_LOCAL_V4I(p)  ((__attribute__((address_space(3))) v4i32_async*)(p))
#endif

__device__ __forceinline__ v8f wmma_bf16(bf16x16 a, bf16x16 b, v8f c) {
  // (neg_a, A, neg_b, B, c_mod, C, reuse_a, reuse_b)
  return __builtin_amdgcn_wmma_f32_16x16x32_bf16(false, a, false, b, (short)0, c,
                                                 false, false);
}

__device__ __forceinline__ v8f zero8() {
  v8f z;
#pragma unroll
  for (int i = 0; i < 8; ++i) z[i] = 0.0f;
  return z;
}

// A-matrix 16x32 bf16 fragment from row-major LDS tile (stride in elements).
// ISA 7.12.2: lane m=lane&15; elems 0..7 -> K=8*half+i, elems 8..15 -> K=16+8*half+i
__device__ __forceinline__ bf16x16 load_a_frag(const __bf16* base, int stride, int lane) {
  const int m = lane & 15, half = lane >> 4;
  const __bf16* p = base + m * stride;
  bf16x16 a;
#pragma unroll
  for (int i = 0; i < 8; ++i) {
    a[i]     = p[8 * half + i];
    a[8 + i] = p[16 + 8 * half + i];
  }
  return a;
}

// B-matrix 32x16 bf16 fragment from LDS tile stored [n][k] (k contiguous).
// lane n=lane&15; elems 0..7 -> K=8*half+i, elems 8..15 -> K=16+8*half+i
__device__ __forceinline__ bf16x16 load_b_frag(const __bf16* base, int stride, int lane) {
  const int n = lane & 15, half = lane >> 4;
  const __bf16* p = base + n * stride;
  bf16x16 b;
#pragma unroll
  for (int i = 0; i < 8; ++i) {
    b[i]     = p[8 * half + i];
    b[8 + i] = p[16 + 8 * half + i];
  }
  return b;
}

// ---------------------------------------------------------------------------
// Generic 64x64-tile bf16 WMMA GEMM with fused fp32->bf16 conversion and
// three epilogues. Block = 128 threads (4 waves); wave w owns rows 16w..16w+15.
// ---------------------------------------------------------------------------
enum { EPI_Q = 0, EPI_KV = 1, EPI_OUT = 2 };

template <int EPI>
__global__ __launch_bounds__(128) void gemm_bf16_kernel(
    const void* __restrict__ Aop, const float* __restrict__ W,
    const float* __restrict__ bias, void* __restrict__ O0, void* __restrict__ O1,
    float* __restrict__ Of, int Mrows, int K, int Ncols) {
  __shared__ __bf16 As[64 * 32];  // [m][k] row-major
  __shared__ __bf16 Bs[64 * 32];  // [n][k] (transposed for B-frag loads)

  const int tid  = threadIdx.x;
  const int lane = tid & 31;
  const int w    = tid >> 5;
  const int row0 = blockIdx.y * 64;
  const int col0 = blockIdx.x * 64;
  const int z    = blockIdx.z;

  const float*  Af = (const float*)Aop + (size_t)z * Mrows * K;
  const __bf16* Ah = (const __bf16*)Aop + (size_t)z * Mrows * K;

  v8f acc[4];
#pragma unroll
  for (int t = 0; t < 4; ++t) acc[t] = zero8();

  for (int kk = 0; kk < K; kk += 32) {
    // A tile 64x32 (coalesced rows of 32 floats)
#pragma unroll
    for (int i = 0; i < 16; ++i) {
      int idx = tid + i * 128;
      int r = idx >> 5, c = idx & 31;
      float v;
      if constexpr (EPI == EPI_OUT) v = (float)Ah[(size_t)(row0 + r) * K + kk + c];
      else                          v = Af[(size_t)(row0 + r) * K + kk + c];
      As[r * 32 + c] = (__bf16)v;
    }
    // B tile 32x64 -> Bs[n][k] (coalesced rows of 64 floats, transposed store)
#pragma unroll
    for (int i = 0; i < 16; ++i) {
      int idx = tid + i * 128;
      int k = idx >> 6, n = idx & 63;
      Bs[n * 32 + k] = (__bf16)W[(size_t)(kk + k) * Ncols + col0 + n];
    }
    if (kk + 32 < K)  // global_prefetch_b8 of next weight tile
      __builtin_prefetch(&W[(size_t)(kk + 32) * Ncols + col0 + lane], 0, 0);
    __syncthreads();

    bf16x16 a = load_a_frag(As + 16 * w * 32, 32, lane);
#pragma unroll
    for (int t = 0; t < 4; ++t) {
      bf16x16 b = load_b_frag(Bs + 16 * t * 32, 32, lane);
      acc[t] = wmma_bf16(a, b, acc[t]);
    }
    __syncthreads();
  }

  const int nlo = lane & 15, half = lane >> 4;
#pragma unroll
  for (int t = 0; t < 4; ++t) {
#pragma unroll
    for (int r = 0; r < 8; ++r) {
      float v = acc[t][r];
      int m = row0 + 16 * w + r + 8 * half;
      int n = col0 + 16 * t + nlo;
      if constexpr (EPI == EPI_Q) {
        // fold softmax scale 64^-0.5 into Q; layout [B*H, M, 64]
        int h = n >> 6, d = n & 63;
        ((__bf16*)O0)[(((size_t)z * 16 + h) * 2048 + m) * 64 + d] = (__bf16)(v * 0.125f);
      } else if constexpr (EPI == EPI_KV) {
        // Wkv columns: [2, H, 64] interleave -> K / V [B*H, N, 64]
        int s = n >> 10, n2 = n & 1023;
        int h = n2 >> 6, d = n2 & 63;
        __bf16* dst = s ? (__bf16*)O1 : (__bf16*)O0;
        dst[(((size_t)z * 16 + h) * 2048 + m) * 64 + d] = (__bf16)v;
      } else {
        Of[(size_t)m * Ncols + n] = v + bias[n];
      }
    }
  }
}

// ---------------------------------------------------------------------------
// Flash-attention core: block = (b,h) x 64 query rows, 4 waves.
// Loop over 64-key tiles: S=Q*K^T (WMMA), online softmax (shfl_xor row
// reductions), P staged through LDS for A-frag re-layout, O += P*V (WMMA).
// Q and K tiles are staged with async global->LDS DMA when available.
// ---------------------------------------------------------------------------
__global__ __launch_bounds__(128) void attn_fwd_kernel(
    const void* __restrict__ Qv, const void* __restrict__ Kv,
    const void* __restrict__ Vv, void* __restrict__ Xv) {
  constexpr int NK = 2048;
  __shared__ __bf16 Qs[64 * 64];  // [m][d]
  __shared__ __bf16 Ks[64 * 64];  // [key][d]  (== column-major of K^T)
  __shared__ __bf16 Vs[64 * 64];  // [d][key]  (transposed for B-frag loads)
  __shared__ __bf16 Ps[64 * 64];  // [m][key]

  const int tid = threadIdx.x, lane = tid & 31, w = tid >> 5;
  const int bh = blockIdx.y;
  const int b = bh >> 4, h = bh & 15;
  const int q0 = blockIdx.x * 64;
  const int nlo = lane & 15, half = lane >> 4;

  const __bf16* Qb = (const __bf16*)Qv;
  const __bf16* Kb = (const __bf16*)Kv;
  const __bf16* Vb = (const __bf16*)Vv;
  __bf16* Xb = (__bf16*)Xv;

  {  // stage Q tile once (8KB contiguous)
    const __bf16* src = Qb + ((size_t)bh * 2048 + q0) * 64;
#if HAVE_ASYNC_LDS
#pragma unroll
    for (int i = 0; i < 4; ++i) {
      int cidx = tid + i * 128;  // 512 x 16B chunks
      __builtin_amdgcn_global_load_async_to_lds_b128(
          AS_GLOBAL_V4I((const char*)src + cidx * 16),
          AS_LOCAL_V4I((char*)Qs + cidx * 16), 0, 0);
    }
    __builtin_amdgcn_s_wait_asynccnt(0);
#else
    const uint32_t* s32 = (const uint32_t*)src;
    uint32_t* d32 = (uint32_t*)Qs;
#pragma unroll
    for (int i = 0; i < 16; ++i) d32[tid + i * 128] = s32[tid + i * 128];
#endif
  }
  __syncthreads();

  bf16x16 qa[2];
#pragma unroll
  for (int c = 0; c < 2; ++c) qa[c] = load_a_frag(Qs + 16 * w * 64 + 32 * c, 64, lane);

  float mrow[8], lrow[8];
  v8f o[4];
#pragma unroll
  for (int r = 0; r < 8; ++r) { mrow[r] = -1e30f; lrow[r] = 0.f; }
#pragma unroll
  for (int t = 0; t < 4; ++t) o[t] = zero8();

  for (int j0 = 0; j0 < NK; j0 += 64) {
    {  // stage K tile (straight copy, async DMA path) ...
      const __bf16* ksrc = Kb + ((size_t)bh * NK + j0) * 64;
#if HAVE_ASYNC_LDS
#pragma unroll
      for (int i = 0; i < 4; ++i) {
        int cidx = tid + i * 128;
        __builtin_amdgcn_global_load_async_to_lds_b128(
            AS_GLOBAL_V4I((const char*)ksrc + cidx * 16),
            AS_LOCAL_V4I((char*)Ks + cidx * 16), 0, 0);
      }
#else
      const uint32_t* s32 = (const uint32_t*)ksrc;
      uint32_t* d32 = (uint32_t*)Ks;
#pragma unroll
      for (int i = 0; i < 16; ++i) d32[tid + i * 128] = s32[tid + i * 128];
#endif
      // ... + V tile transposed into [d][key] (must reshape -> VALU copy)
      const uint16_t* vsrc = (const uint16_t*)(Vb + ((size_t)bh * NK + j0) * 64);
      uint16_t* vdst = (uint16_t*)Vs;
#pragma unroll
      for (int i = 0; i < 32; ++i) {
        int idx = tid + i * 128;
        int key = idx >> 6, d = idx & 63;
        vdst[d * 64 + key] = vsrc[idx];
      }
#if HAVE_ASYNC_LDS
      __builtin_amdgcn_s_wait_asynccnt(0);
#endif
    }
    __syncthreads();

    // S tile: 16 rows x 64 keys per wave, hd=64 -> 2 K-chunks per col-tile
    v8f sf[4];
#pragma unroll
    for (int t = 0; t < 4; ++t) {
      v8f s = zero8();
#pragma unroll
      for (int c = 0; c < 2; ++c) {
        bf16x16 kb = load_b_frag(Ks + 16 * t * 64 + 32 * c, 64, lane);
        s = wmma_bf16(qa[c], kb, s);
      }
      sf[t] = s;
    }

    // online softmax: row stats live per-lane for rows r+8*half
#pragma unroll
    for (int r = 0; r < 8; ++r) {
      float x = fmaxf(fmaxf(sf[0][r], sf[1][r]), fmaxf(sf[2][r], sf[3][r]));
      x = fmaxf(x, __shfl_xor(x, 1, 32));
      x = fmaxf(x, __shfl_xor(x, 2, 32));
      x = fmaxf(x, __shfl_xor(x, 4, 32));
      x = fmaxf(x, __shfl_xor(x, 8, 32));
      float mn = fmaxf(mrow[r], x);
      float alpha = __expf(mrow[r] - mn);
      mrow[r] = mn;
      float ps = 0.f;
#pragma unroll
      for (int t = 0; t < 4; ++t) {
        float p = __expf(sf[t][r] - mn);
        sf[t][r] = p;
        ps += p;
      }
      ps += __shfl_xor(ps, 1, 32);
      ps += __shfl_xor(ps, 2, 32);
      ps += __shfl_xor(ps, 4, 32);
      ps += __shfl_xor(ps, 8, 32);
      lrow[r] = lrow[r] * alpha + ps;
#pragma unroll
      for (int t = 0; t < 4; ++t) o[t][r] *= alpha;
    }

    // re-layout P through LDS (C-frag layout -> row-major [m][key])
#pragma unroll
    for (int t = 0; t < 4; ++t)
#pragma unroll
      for (int r = 0; r < 8; ++r)
        Ps[(16 * w + r + 8 * half) * 64 + 16 * t + nlo] = (__bf16)sf[t][r];
    __syncthreads();

    // O += P x V  (K dim = 64 keys -> 2 chunks; 4 col-tiles of hd)
#pragma unroll
    for (int c = 0; c < 2; ++c) {
      bf16x16 pa = load_a_frag(Ps + 16 * w * 64 + 32 * c, 64, lane);
#pragma unroll
      for (int t = 0; t < 4; ++t) {
        bf16x16 vb = load_b_frag(Vs + 16 * t * 64 + 32 * c, 64, lane);
        o[t] = wmma_bf16(pa, vb, o[t]);
      }
    }
    __syncthreads();
  }

  // epilogue: normalize and scatter to X [B*M, C] bf16 (head-major columns)
#pragma unroll
  for (int t = 0; t < 4; ++t)
#pragma unroll
    for (int r = 0; r < 8; ++r) {
      int m = q0 + 16 * w + r + 8 * half;
      float v = o[t][r] / lrow[r];
      Xb[((size_t)b * 2048 + m) * 1024 + h * 64 + 16 * t + nlo] = (__bf16)v;
    }
}

// ---------------------------------------------------------------------------
// Host launch. Workspace: Qb(8MB) Kb(8MB) Vb(8MB) Xb(8MB), all bf16.
// ---------------------------------------------------------------------------
extern "C" void kernel_launch(void* const* d_in, const int* in_sizes, int n_in,
                              void* d_out, int out_size, void* d_ws, size_t ws_size,
                              hipStream_t stream) {
  (void)in_sizes; (void)n_in; (void)out_size; (void)ws_size;
  const float* ref  = (const float*)d_in[0];  // [2,2048,1024]
  const float* tgt  = (const float*)d_in[1];  // [2,2048,1024]
  const float* Wq   = (const float*)d_in[2];  // [1024,1024]
  const float* Wkv  = (const float*)d_in[3];  // [1024,2048]
  const float* Wprj = (const float*)d_in[4];  // [1024,1024]
  const float* bprj = (const float*)d_in[5];  // [1024]
  float* out = (float*)d_out;                 // [2,2048,1024]

  char* ws = (char*)d_ws;
  void* Qb = ws;
  void* Kb = ws + (size_t)8  * 1024 * 1024;
  void* Vb = ws + (size_t)16 * 1024 * 1024;
  void* Xb = ws + (size_t)24 * 1024 * 1024;

  dim3 blk(128);
  // Q = tgt @ Wq  (per batch): cols 1024 -> 16 tiles, rows 2048 -> 32 tiles
  gemm_bf16_kernel<EPI_Q><<<dim3(16, 32, 2), blk, 0, stream>>>(
      tgt, Wq, nullptr, Qb, nullptr, nullptr, 2048, 1024, 1024);
  // KV = ref @ Wkv: cols 2048 -> 32 tiles
  gemm_bf16_kernel<EPI_KV><<<dim3(32, 32, 2), blk, 0, stream>>>(
      ref, Wkv, nullptr, Kb, Vb, nullptr, 2048, 1024, 2048);
  // attention: 32 query tiles x (B*H = 32)
  attn_fwd_kernel<<<dim3(32, 32), blk, 0, stream>>>(Qb, Kb, Vb, Xb);
  // out = X @ Wproj + b: rows B*M = 4096 -> 64 tiles
  gemm_bf16_kernel<EPI_OUT><<<dim3(16, 64, 1), blk, 0, stream>>>(
      Xb, Wprj, bprj, nullptr, nullptr, out, 4096, 1024, 1024);
}
